// EEGSNNRouteC_50414326120528
// MI455X (gfx1250) — matrix operand order repaired and checked by reading, not compiled, not run on hardware
//
#include <hip/hip_runtime.h>
#include <hip/hip_bf16.h>
#include <cstddef>

// ---------------------------------------------------------------------------
// EEG SNN pipeline for MI455X (gfx1250, wave32, WMMA).
//   x:(256,9,384,14) f32, W1:(256,126), W2:(128,256), W3:(64,128),
//   head_w:(4,64), head_b:(4)  ->  logits:(256,4) f32
// delta-pack(fragment-major) -> f32 WMMA GEMM (LDS-staged W fragments) ->
// CUBA scan -> f16 WMMA GEMM -> scan -> f16 WMMA GEMM -> scan+mean -> head
// ---------------------------------------------------------------------------

typedef _Float16 v16h __attribute__((ext_vector_type(16)));
typedef _Float16 v8h  __attribute__((ext_vector_type(8)));
typedef float    v8f  __attribute__((ext_vector_type(8)));
typedef float    v4f  __attribute__((ext_vector_type(4)));
typedef float    v2f  __attribute__((ext_vector_type(2)));

static constexpr int B_   = 256;
static constexpr int T_   = 384;
static constexpr int MTOT = T_ * B_;     // 98304 rows (6144 M-tiles)
static constexpr int K1   = 128;         // 126 padded to 128
static constexpr int H1   = 256;
static constexpr int H2   = 128;
static constexpr int H3   = 64;

// f32 WMMA fragment packing (16x16x4): a lane's fragment for K-step ks is
// {X[row][4*ks + 2*half], X[row][4*ks + 2*half + 1]}, half = lane>>4.
// We pack TWO consecutive K-steps per lane as one float4 ("kpair" p):
//   e0,e1 -> i = 8p + 2*half + {0,1};  e2,e3 -> i = 8p + 4 + 2*half + {0,1}
__device__ __forceinline__ int frag_i(int p, int lane, int e) {
    return 8 * p + 2 * (lane >> 4) + 4 * (e >> 1) + (e & 1);
}

// ---------------------------------------------------------------------------
// W1 -> fragment-major packed Wf: idx = (((g*4 + j)*16 + p)*32 + lane)*4 + e
//   n = g*64 + j*16 + (lane&15), i = frag_i(p,lane,e), zero-pad i >= 126
// ---------------------------------------------------------------------------
__global__ __launch_bounds__(256) void pack_w1_kernel(const float* __restrict__ W1,
                                                      float* __restrict__ Wf) {
    int idx  = blockIdx.x * blockDim.x + threadIdx.x;   // 32768
    int e    = idx & 3;
    int lane = (idx >> 2) & 31;
    int p    = (idx >> 7) & 15;
    int j    = (idx >> 11) & 3;
    int g    = idx >> 13;
    int n = g * 64 + j * 16 + (lane & 15);
    int i = frag_i(p, lane, e);
    Wf[idx] = (i < 126) ? W1[n * 126 + i] : 0.0f;
}

__global__ void cvt_f16_kernel(const float* __restrict__ W, _Float16* __restrict__ Wh, int n) {
    int idx = blockIdx.x * blockDim.x + threadIdx.x;
    if (idx < n) Wh[idx] = (_Float16)W[idx];
}

// ---------------------------------------------------------------------------
// Delta encoding + transpose + K-pad, written fragment-major:
//   Af[((tile*16 + p)*32 + lane)*4 + e], row m = tile*16 + (lane&15) = t*B + b,
//   feature i = frag_i(p,lane,e) = c*14 + h (zero for i >= 126)
// ---------------------------------------------------------------------------
__global__ __launch_bounds__(256) void delta_pack_kernel(const float* __restrict__ x,
                                                         float* __restrict__ Af) {
    size_t idx = (size_t)blockIdx.x * blockDim.x + threadIdx.x;  // MTOT*128
    int e    = (int)(idx & 3);
    int lane = (int)((idx >> 2) & 31);
    int p    = (int)((idx >> 7) & 15);
    int tile = (int)(idx >> 11);
    int m = tile * 16 + (lane & 15);
    int b = m & (B_ - 1);
    int t = m >> 8;
    int i = frag_i(p, lane, e);
    float v = 0.0f;
    if (i < 126) {
        int c = i / 14, h = i % 14;
        size_t base = (((size_t)b * 9 + c) * T_ + t) * 14 + h;   // x[b][c][t][h]
        float cur  = x[base];
        float prev = (t > 0) ? x[base - 14] : 0.0f;
        v = cur - prev;
    }
    Af[idx] = v;
}

// ---------------------------------------------------------------------------
// GEMM1 (fp32 WMMA 16x16x4): Z[m][n] = sum_k A[m][k] * W1[n][k]
// Block = 16 M-tiles x one 64-col N-group; 8 waves, each wave MT=2 x NT=4.
// W fragments (32 KB) staged once per block into LDS ([j][p][lane] float4,
// conflict-free ds_load_b128); A read as coalesced b128 (one kpair per load).
// Per kpair: 2 global b128 + 4 LDS b128 feed 16 wmma.
// ---------------------------------------------------------------------------
__global__ __launch_bounds__(256) void gemm_f32_kernel(const float* __restrict__ Af,
                                                       const float* __restrict__ Wf,
                                                       float* __restrict__ C) {
    constexpr int N = H1;
    __shared__ v4f lw[2048];                 // 32 KB W fragments for this N-group
    const int g    = blockIdx.x & 3;
    const int mblk = blockIdx.x >> 2;
    const v4f* wsrc = (const v4f*)Wf + (size_t)g * 2048;
    #pragma unroll
    for (int i = threadIdx.x; i < 2048; i += 256) lw[i] = wsrc[i];
    __syncthreads();

    const int wave = threadIdx.x >> 5;
    const int lane = threadIdx.x & 31;
    const int tile0 = mblk * 16 + wave * 2;  // this wave's 2 M-tiles

    v8f acc[2][4] = {};
    const v4f* a0 = (const v4f*)Af + (size_t)tile0 * 512 + lane;  // 512 v4f per tile
    const v4f* a1 = a0 + 512;
    #pragma unroll 4
    for (int p = 0; p < 16; ++p) {
        v4f fa0 = a0[p * 32];
        v4f fa1 = a1[p * 32];
        v2f a0lo = __builtin_shufflevector(fa0, fa0, 0, 1);
        v2f a0hi = __builtin_shufflevector(fa0, fa0, 2, 3);
        v2f a1lo = __builtin_shufflevector(fa1, fa1, 0, 1);
        v2f a1hi = __builtin_shufflevector(fa1, fa1, 2, 3);
        #pragma unroll
        for (int j = 0; j < 4; ++j) {
            v4f fw = lw[(j * 16 + p) * 32 + lane];
            v2f wlo = __builtin_shufflevector(fw, fw, 0, 1);
            v2f whi = __builtin_shufflevector(fw, fw, 2, 3);
            acc[0][j] = __builtin_amdgcn_wmma_f32_16x16x4_f32(false, a0lo, false, wlo, (short)0, acc[0][j], false, false);
            acc[0][j] = __builtin_amdgcn_wmma_f32_16x16x4_f32(false, a0hi, false, whi, (short)0, acc[0][j], false, false);
            acc[1][j] = __builtin_amdgcn_wmma_f32_16x16x4_f32(false, a1lo, false, wlo, (short)0, acc[1][j], false, false);
            acc[1][j] = __builtin_amdgcn_wmma_f32_16x16x4_f32(false, a1hi, false, whi, (short)0, acc[1][j], false, false);
        }
    }
    const int l = lane & 15, half = lane >> 4, n0 = g * 64;
    #pragma unroll
    for (int mt = 0; mt < 2; ++mt)
        #pragma unroll
        for (int j = 0; j < 4; ++j)
            #pragma unroll
            for (int r = 0; r < 8; ++r)   // C f32 16x16: VGPR r -> row M = r + 8*half
                C[(size_t)((tile0 + mt) * 16 + r + 8 * half) * N + n0 + j * 16 + l] = acc[mt][j][r];
}

// ---------------------------------------------------------------------------
// GEMM f16 (WMMA 16x16x32, f32 accum): Z[m][n] = sum_k A[m][k] * W[n][k]
// Wave = MT=2 M-tiles x NT=4 N-tiles. A 16x32 layout per lane: chunks
// [k0+8*half,+8) and [k0+16+8*half,+8); B 32x16 per lane: column n=lane&15,
// same K chunks [k0+8*half,+8) and [k0+16+8*half,+8) of W row n -> 2x b128 each.
// ---------------------------------------------------------------------------
template <int K, int N>
__global__ __launch_bounds__(256) void gemm_f16_kernel(const _Float16* __restrict__ A,
                                                       const _Float16* __restrict__ W,
                                                       float* __restrict__ C) {
    constexpr int NGROUPS = N / 64;
    const int wave = (blockIdx.x * blockDim.x + threadIdx.x) >> 5;
    const int lane = threadIdx.x & 31;
    const int m0 = (wave / NGROUPS) * 32;
    const int n0 = (wave % NGROUPS) * 64;
    const int l    = lane & 15;
    const int half = lane >> 4;

    v8f acc[2][4] = {};
    const _Float16* arow0 = A + (size_t)(m0 + l) * K + 8 * half;
    const _Float16* arow1 = arow0 + (size_t)16 * K;
    #pragma unroll
    for (int k0 = 0; k0 < K; k0 += 32) {
        v16h a[2];
        {
            v8h lo0 = *(const v8h*)(arow0 + k0);
            v8h hi0 = *(const v8h*)(arow0 + k0 + 16);
            v8h lo1 = *(const v8h*)(arow1 + k0);
            v8h hi1 = *(const v8h*)(arow1 + k0 + 16);
            #pragma unroll
            for (int i = 0; i < 8; ++i) {
                a[0][i] = lo0[i]; a[0][8 + i] = hi0[i];
                a[1][i] = lo1[i]; a[1][8 + i] = hi1[i];
            }
        }
        #pragma unroll
        for (int j = 0; j < 4; ++j) {
            const _Float16* wrow = W + (size_t)(n0 + j * 16 + l) * K + 8 * half + k0;
            v8h blo = *(const v8h*)(wrow);
            v8h bhi = *(const v8h*)(wrow + 16);
            v16h bm;
            #pragma unroll
            for (int i = 0; i < 8; ++i) { bm[i] = blo[i]; bm[8 + i] = bhi[i]; }
            acc[0][j] = __builtin_amdgcn_wmma_f32_16x16x32_f16(false, a[0], false, bm, (short)0, acc[0][j], false, false);
            acc[1][j] = __builtin_amdgcn_wmma_f32_16x16x32_f16(false, a[1], false, bm, (short)0, acc[1][j], false, false);
        }
    }
    #pragma unroll
    for (int mt = 0; mt < 2; ++mt)
        #pragma unroll
        for (int j = 0; j < 4; ++j)
            #pragma unroll
            for (int r = 0; r < 8; ++r)
                C[(size_t)(m0 + mt * 16 + r + 8 * half) * N + n0 + j * 16 + l] = acc[mt][j][r];
}

// ---------------------------------------------------------------------------
// CUBA LIF scan over T. One thread per (b, n); coalesced over n each step.
// WRITE=true: emit f16 spikes (exact 0/1). WRITE=false: fused T-mean -> feat.
// ---------------------------------------------------------------------------
template <bool WRITE>
__global__ __launch_bounds__(256) void cuba_kernel(const float* __restrict__ Z,
                                                   _Float16* __restrict__ S,
                                                   float* __restrict__ feat, int N) {
    const int idx = blockIdx.x * blockDim.x + threadIdx.x;   // b*N + n
    const int n = idx % N;
    const int b = idx / N;
    float cur = 0.0f, vol = 0.0f, cnt = 0.0f;
    for (int t = 0; t < T_; ++t) {
        size_t off = ((size_t)t * B_ + b) * N + n;
        float z = Z[off];
        cur = 0.75f * cur + z;          // (1 - CUR_DECAY)
        vol = 0.97f * vol + cur;        // (1 - VOLT_DECAY)
        float s = (vol >= 1.0f) ? 1.0f : 0.0f;
        vol *= (1.0f - s);              // hard reset
        if (WRITE) S[off] = (_Float16)s;
        else       cnt += s;
    }
    if (!WRITE) feat[idx] = cnt * (1.0f / (float)T_);
}

// ---------------------------------------------------------------------------
// Head: logits[b][c] = feat[b] . head_w[c] + head_b[c]
// ---------------------------------------------------------------------------
__global__ __launch_bounds__(256) void head_kernel(const float* __restrict__ feat,
                                                   const float* __restrict__ hw,
                                                   const float* __restrict__ hb,
                                                   float* __restrict__ out) {
    int idx = blockIdx.x * blockDim.x + threadIdx.x;  // 1024
    if (idx >= B_ * 4) return;
    int c = idx & 3, b = idx >> 2;
    float acc = hb[c];
    #pragma unroll
    for (int i = 0; i < H3; ++i) acc += feat[b * H3 + i] * hw[c * H3 + i];
    out[idx] = acc;
}

// ---------------------------------------------------------------------------
extern "C" void kernel_launch(void* const* d_in, const int* in_sizes, int n_in,
                              void* d_out, int out_size, void* d_ws, size_t ws_size,
                              hipStream_t stream) {
    const float* x      = (const float*)d_in[0];
    const float* W1     = (const float*)d_in[1];
    const float* W2     = (const float*)d_in[2];
    const float* W3     = (const float*)d_in[3];
    const float* head_w = (const float*)d_in[4];
    const float* head_b = (const float*)d_in[5];
    float* logits = (float*)d_out;

    // Workspace layout (~202 MB):
    //   [0, 50.3MB)        Af (fragment-packed delta, f32) -- reused as S2 later
    //   [50.3, 151.0MB)    Z  (f32, reused for Z1/Z2/Z3)
    //   [151.0, 201.3MB)   S1 (f16 spikes)
    //   [201.3MB, ...)     Wf / W2h / W3h / feat
    char* ws = (char*)d_ws;
    const size_t AF_BYTES = (size_t)MTOT * K1 * 4;          // 50331648
    const size_t Z_BYTES  = (size_t)MTOT * H1 * 4;          // 100663296
    const size_t S1_BYTES = (size_t)MTOT * H1 * 2;          // 50331648
    float*    Af  = (float*)ws;
    _Float16* S2  = (_Float16*)ws;                          // aliases Af (dead after GEMM1)
    float*    Z   = (float*)(ws + AF_BYTES);
    _Float16* S1  = (_Float16*)(ws + AF_BYTES + Z_BYTES);
    char*     sm  = ws + AF_BYTES + Z_BYTES + S1_BYTES;
    float*    Wf  = (float*)sm;                             // 256*128*4
    _Float16* W2h = (_Float16*)(sm + 131072);               // 128*256*2
    _Float16* W3h = (_Float16*)(sm + 131072 + 65536);       // 64*128*2
    float*    feat = (float*)(sm + 131072 + 65536 + 16384); // 256*64*4
    (void)in_sizes; (void)n_in; (void)out_size; (void)ws_size;

    // 1) weight prep
    pack_w1_kernel<<<(H1 * K1) / 256, 256, 0, stream>>>(W1, Wf);
    cvt_f16_kernel<<<(H2 * H1) / 256, 256, 0, stream>>>(W2, W2h, H2 * H1);
    cvt_f16_kernel<<<(H3 * H2) / 256, 256, 0, stream>>>(W3, W3h, H3 * H2);

    // 2) delta encoding into fragment-packed Af
    delta_pack_kernel<<<((size_t)MTOT * K1) / 256, 256, 0, stream>>>(x, Af);

    // 3) layer 1: fp32 WMMA GEMM. 384 M-blocks x 4 N-groups = 1536 blocks.
    gemm_f32_kernel<<<(MTOT / 256) * 4, 256, 0, stream>>>(Af, Wf, Z);
    cuba_kernel<true><<<(B_ * H1) / 256, 256, 0, stream>>>(Z, S1, nullptr, H1);

    // 4) layer 2: f16 WMMA GEMM (spikes exact in f16). 6144 waves -> 768 blocks.
    gemm_f16_kernel<H1, H2><<<(MTOT / 32) * 2 / 8, 256, 0, stream>>>(S1, W2h, Z);
    cuba_kernel<true><<<(B_ * H2) / 256, 256, 0, stream>>>(Z, S2, nullptr, H2);

    // 5) layer 3: f16 WMMA GEMM + scan fused with T-mean. 3072 waves -> 384 blocks.
    gemm_f16_kernel<H2, H3><<<(MTOT / 32) * 1 / 8, 256, 0, stream>>>(S2, W3h, Z);
    cuba_kernel<false><<<(B_ * H3) / 256, 256, 0, stream>>>(Z, nullptr, feat, H3);

    // 6) classifier head
    head_kernel<<<4, 256, 0, stream>>>(feat, head_w, head_b, logits);
}
